// SpookyNetEmbedding_21414706938519
// MI455X (gfx1250) — matrix operand order repaired and checked by reading, not compiled, not run on hardware
//
#include <hip/hip_runtime.h>
#include <hip/hip_bf16.h>
#include <math.h>

// ---------------------------------------------------------------------------
// SpookyNet embedding forward, CDNA5 (gfx1250), wave32 + WMMA bf16 tensor ops.
// N=8192 nodes, E=131072 edges, DIM=128, NBASIS=16, NLAYERS=3.
//
//  - v_wmma_f32_16x16x32_bf16 for all dense GEMMs (fp32 in memory, bf16 tiles,
//    f32 accumulation).
//  - Fully fused resMLP kernel: 3 chained 128x128 GEMMs per kernel, weights
//    staged as transposed bf16 in LDS (96 KB) + wave-private activation slab
//    (32 KB) -> ~130 KB LDS per workgroup (needs CDNA5's 320 KB/WGP LDS).
//  - SiLU via v_rcp_f32 (not IEEE divide), packed bf16 weight staging.
// ---------------------------------------------------------------------------

typedef __attribute__((ext_vector_type(16))) __bf16 v16bf;
typedef __attribute__((ext_vector_type(8)))  __bf16 v8bf;
typedef __attribute__((ext_vector_type(2)))  __bf16 v2bf;
typedef __attribute__((ext_vector_type(8)))  float  v8f;

#define DIMF 128
#define NB   16

__device__ __forceinline__ float silu_f(float x) {
    // x * sigmoid(x) via v_exp_f32 + v_rcp_f32 (avoids IEEE div chain)
    return x * __builtin_amdgcn_rcpf(1.0f + __expf(-x));
}

// ---- Stage W[128 x Nout] (fp32 row-major) -> wt[Nout][128] bf16 (transposed)
// Each thread handles a 2(k) x 4(n) chunk: two coalesced float4 loads, four
// packed-bf16 ds_store_b32 (pair = {W[2k][n], W[2k+1][n]} = wt[n][2k..2k+1]).
__device__ __forceinline__ void stage_weights_bf16t(
    __bf16* __restrict__ wt, const float* __restrict__ W, int Nout, int tid)
{
    const int nq     = Nout >> 2;         // n-quads per k-row
    const int chunks = 64 * nq;           // (128/2) * (Nout/4)
    for (int c = tid; c < chunks; c += 256) {
        const int kk = c / nq;            // 0..63  -> k = 2*kk
        const int n0 = (c - kk * nq) * 4; // 0..Nout-4
        const float4 w0 = *(const float4*)(W + (size_t)(2 * kk)     * Nout + n0);
        const float4 w1 = *(const float4*)(W + (size_t)(2 * kk + 1) * Nout + n0);
        const float a0[4] = {w0.x, w0.y, w0.z, w0.w};
        const float a1[4] = {w1.x, w1.y, w1.z, w1.w};
        #pragma unroll
        for (int j = 0; j < 4; ++j) {
            v2bf p;
            p[0] = (__bf16)a0[j];
            p[1] = (__bf16)a1[j];
            *(v2bf*)(&wt[(n0 + j) * 128 + 2 * kk]) = p;
        }
    }
}

// ---- A fragment (16x32 bf16, ISA layout) from fp32 row pointer -------------
// lanes 0-15:  K = c0 + {0..7} and c0 + {16..23}   (c0 = ks*32 + khalf*8)
// lanes 16-31: same formula with khalf = 1
__device__ __forceinline__ v16bf make_a_frag_global(
    const float* __restrict__ Xrow, const float* __restrict__ X2row,
    int c0, bool act)
{
    float xa[16];
    {
        const float4 p0 = *(const float4*)(Xrow + c0);
        const float4 p1 = *(const float4*)(Xrow + c0 + 4);
        const float4 p2 = *(const float4*)(Xrow + c0 + 16);
        const float4 p3 = *(const float4*)(Xrow + c0 + 20);
        xa[0] = p0.x;  xa[1] = p0.y;  xa[2]  = p0.z;  xa[3]  = p0.w;
        xa[4] = p1.x;  xa[5] = p1.y;  xa[6]  = p1.z;  xa[7]  = p1.w;
        xa[8] = p2.x;  xa[9] = p2.y;  xa[10] = p2.z;  xa[11] = p2.w;
        xa[12]= p3.x;  xa[13]= p3.y;  xa[14] = p3.z;  xa[15] = p3.w;
    }
    if (X2row) {
        const float4 q0 = *(const float4*)(X2row + c0);
        const float4 q1 = *(const float4*)(X2row + c0 + 4);
        const float4 q2 = *(const float4*)(X2row + c0 + 16);
        const float4 q3 = *(const float4*)(X2row + c0 + 20);
        xa[0] += q0.x;  xa[1] += q0.y;  xa[2]  += q0.z;  xa[3]  += q0.w;
        xa[4] += q1.x;  xa[5] += q1.y;  xa[6]  += q1.z;  xa[7]  += q1.w;
        xa[8] += q2.x;  xa[9] += q2.y;  xa[10] += q2.z;  xa[11] += q2.w;
        xa[12]+= q3.x;  xa[13]+= q3.y;  xa[14] += q3.z;  xa[15] += q3.w;
    }
    if (act) {
        #pragma unroll
        for (int e = 0; e < 16; ++e) xa[e] = silu_f(xa[e]);
    }
    v16bf a;
    #pragma unroll
    for (int e = 0; e < 16; ++e) a[e] = (__bf16)xa[e];
    return a;
}

// ---- A fragment from bf16 LDS activation slab (already silu'd) -------------
__device__ __forceinline__ v16bf make_a_frag_lds(const __bf16* rowp, int c0)
{
    const v8bf lo = *(const v8bf*)(rowp + c0);
    const v8bf hi = *(const v8bf*)(rowp + c0 + 16);
    v16bf a;
    #pragma unroll
    for (int e = 0; e < 8; ++e) { a[e] = lo[e]; a[8 + e] = hi[e]; }
    return a;
}

// ---------------------------------------------------------------------------
// Fused resMLP:    h  = silu(x) @ W1 + b1
//                  rb = x + silu(h) @ W2 + b2        (x = X + X2)
//   resOnly:       Out = rb                           (flags bit2)
//   else:          Out (+)= silu(rb) @ Wo + bo       (accumulate: flags bit1)
//
// Block: 256 threads = 8 waves; each wave owns 16 rows x 128 cols (8 WMMA
// tiles, 64 acc VGPRs). Activation slab in LDS is wave-private (each wave
// writes/reads only its own 16 rows) so no barriers after weight staging.
// ---------------------------------------------------------------------------
__global__ __launch_bounds__(256) void resmlp_fused(
    const float* __restrict__ X,  const float* __restrict__ X2,
    const float* __restrict__ W1, const float* __restrict__ b1,
    const float* __restrict__ W2, const float* __restrict__ b2,
    const float* __restrict__ Wo, const float* __restrict__ bo,
    float* __restrict__ Out, int M, int flags)
{
    extern __shared__ __align__(32) char smem[];
    __bf16* wt1  = (__bf16*)smem;            // [128][128] transposed (n-major)
    __bf16* wt2  = wt1 + 128 * 128;
    __bf16* wto  = wt2 + 128 * 128;
    __bf16* actb = wto + 128 * 128;          // [128 rows][128] bf16 acts
    float*  sb1  = (float*)(actb + 128 * 128);
    float*  sb2  = sb1 + 128;
    float*  sbo  = sb2 + 128;

    const int  tid     = threadIdx.x;
    const bool accum   = (flags & 2) != 0;
    const bool resOnly = (flags & 4) != 0;

    stage_weights_bf16t(wt1, W1, 128, tid);
    stage_weights_bf16t(wt2, W2, 128, tid);
    if (!resOnly) stage_weights_bf16t(wto, Wo, 128, tid);
    if (tid < 128) {
        sb1[tid] = b1[tid];
        sb2[tid] = b2[tid];
        sbo[tid] = resOnly ? 0.0f : bo[tid];
    }
    __syncthreads();

    const int wave  = tid >> 5;
    const int lane  = tid & 31;
    const int nl    = lane & 15;
    const int khalf = lane >> 4;
    const int m0    = blockIdx.x * 128 + wave * 16;   // global row base of wave
    const int mrow  = m0 + nl;                        // A-fragment row
    const int lrow  = wave * 16 + nl;                 // LDS-slab A row (local)
    const int mbase = m0 + ((lane < 16) ? 0 : 8);     // C-tile row base
    const int lbase = wave * 16 + ((lane < 16) ? 0 : 8);

    const float* Xrow  = X + (size_t)mrow * 128;
    const float* X2row = X2 ? (X2 + (size_t)mrow * 128) : nullptr;

    v8f acc[8];

    // ---------------- stage 1: h = silu(x) @ W1 + b1 ----------------
    #pragma unroll
    for (int t = 0; t < 8; ++t)
        #pragma unroll
        for (int r = 0; r < 8; ++r) acc[t][r] = 0.0f;

    #pragma unroll
    for (int ks = 0; ks < 4; ++ks) {
        const int c0 = ks * 32 + khalf * 8;
        const v16bf a = make_a_frag_global(Xrow, X2row, c0, true);
        #pragma unroll
        for (int t = 0; t < 8; ++t) {
            const int n = t * 16 + nl;
            const v16bf b = make_a_frag_lds(&wt1[n * 128], c0);
            acc[t] = __builtin_amdgcn_wmma_f32_16x16x32_bf16(
                false, a, false, b, (short)0, acc[t], false, false);
        }
    }
    // epilogue 1: actb = bf16(silu(h))   (wave-private rows)
    #pragma unroll
    for (int t = 0; t < 8; ++t) {
        const int n = t * 16 + nl;
        const float bv = sb1[n];
        #pragma unroll
        for (int r = 0; r < 8; ++r)
            actb[(lbase + r) * 128 + n] = (__bf16)silu_f(acc[t][r] + bv);
    }

    // ---------------- stage 2: rb = x + silu(h) @ W2 + b2 ----------------
    #pragma unroll
    for (int t = 0; t < 8; ++t)
        #pragma unroll
        for (int r = 0; r < 8; ++r) acc[t][r] = 0.0f;

    #pragma unroll
    for (int ks = 0; ks < 4; ++ks) {
        const int c0 = ks * 32 + khalf * 8;
        const v16bf a = make_a_frag_lds(&actb[lrow * 128], c0);
        #pragma unroll
        for (int t = 0; t < 8; ++t) {
            const int n = t * 16 + nl;
            const v16bf b = make_a_frag_lds(&wt2[n * 128], c0);
            acc[t] = __builtin_amdgcn_wmma_f32_16x16x32_bf16(
                false, a, false, b, (short)0, acc[t], false, false);
        }
    }
    // epilogue 2: add residual x (+x2); either emit rb (resOnly) or stage
    // bf16(silu(rb)) back into the activation slab.
    #pragma unroll
    for (int t = 0; t < 8; ++t) {
        const int n = t * 16 + nl;
        const float bv = sb2[n];
        #pragma unroll
        for (int r = 0; r < 8; ++r) {
            const int m = mbase + r;
            float v = acc[t][r] + bv + X[(size_t)m * 128 + n];
            if (X2) v += X2[(size_t)m * 128 + n];
            if (resOnly) Out[(size_t)m * 128 + n] = v;
            else         actb[(lbase + r) * 128 + n] = (__bf16)silu_f(v);
        }
    }
    if (resOnly) return;

    // ---------------- stage 3: Out (+)= silu(rb) @ Wo + bo ----------------
    #pragma unroll
    for (int t = 0; t < 8; ++t)
        #pragma unroll
        for (int r = 0; r < 8; ++r) acc[t][r] = 0.0f;

    #pragma unroll
    for (int ks = 0; ks < 4; ++ks) {
        const int c0 = ks * 32 + khalf * 8;
        const v16bf a = make_a_frag_lds(&actb[lrow * 128], c0);
        #pragma unroll
        for (int t = 0; t < 8; ++t) {
            const int n = t * 16 + nl;
            const v16bf b = make_a_frag_lds(&wto[n * 128], c0);
            acc[t] = __builtin_amdgcn_wmma_f32_16x16x32_bf16(
                false, a, false, b, (short)0, acc[t], false, false);
        }
    }
    #pragma unroll
    for (int t = 0; t < 8; ++t) {
        const int n = t * 16 + nl;
        const float bv = sbo[n];
        #pragma unroll
        for (int r = 0; r < 8; ++r) {
            const int m = mbase + r;
            const size_t o = (size_t)m * 128 + n;
            float v = acc[t][r] + bv;
            if (accum) v += Out[o];
            Out[o] = v;
        }
    }
}

// ---------------------------------------------------------------------------
// Generic single GEMM (used for P12/D12: Nout=256, no act/bias/residual).
// Out[M x Nout] = X @ W[128 x Nout].  colGroups = Nout/128.
// ---------------------------------------------------------------------------
__global__ __launch_bounds__(256) void gemm128_wmma(
    const float* __restrict__ X, const float* __restrict__ W,
    float* __restrict__ Out, int M, int colGroups)
{
    extern __shared__ __align__(32) char smem[];
    __bf16* wt = (__bf16*)smem;                       // [Nout][128] transposed

    const int Nout = colGroups * 128;
    const int tid  = threadIdx.x;

    stage_weights_bf16t(wt, W, Nout, tid);
    __syncthreads();

    const int wave = tid >> 5;
    const int lane = tid & 31;
    const int rowTilesPerBlock = 8 / colGroups;
    const int rowTile  = wave % rowTilesPerBlock;
    const int colGroup = wave / rowTilesPerBlock;
    const int m0    = blockIdx.x * (rowTilesPerBlock * 16) + rowTile * 16;
    const int nl    = lane & 15;
    const int khalf = lane >> 4;
    const float* Xrow = X + (size_t)(m0 + nl) * 128;

    v8f acc[8];
    #pragma unroll
    for (int t = 0; t < 8; ++t)
        #pragma unroll
        for (int r = 0; r < 8; ++r) acc[t][r] = 0.0f;

    #pragma unroll
    for (int ks = 0; ks < 4; ++ks) {
        const int c0 = ks * 32 + khalf * 8;
        const v16bf a = make_a_frag_global(Xrow, nullptr, c0, false);
        #pragma unroll
        for (int t = 0; t < 8; ++t) {
            const int n = colGroup * 128 + t * 16 + nl;
            const v16bf b = make_a_frag_lds(&wt[n * 128], c0);
            acc[t] = __builtin_amdgcn_wmma_f32_16x16x32_bf16(
                false, a, false, b, (short)0, acc[t], false, false);
        }
    }

    const int mbase = m0 + ((lane < 16) ? 0 : 8);
    #pragma unroll
    for (int t = 0; t < 8; ++t) {
        const int n = colGroup * 128 + t * 16 + nl;
        #pragma unroll
        for (int r = 0; r < 8; ++r)
            Out[(size_t)(mbase + r) * Nout + n] = acc[t][r];
    }
}

// ---------------------------------------------------------------------------
// Node embedding: xi[n,d] = enc_table[s] @ W_species[:,d] + zrand_table[s,d]
// ---------------------------------------------------------------------------
__global__ __launch_bounds__(128) void embed_kernel(
    const int* __restrict__ species, const float* __restrict__ enc,
    const float* __restrict__ zrand, const float* __restrict__ Wsp,
    float* __restrict__ xi, int N)
{
    __shared__ float sw[20 * 128];
    const int d = threadIdx.x;
    for (int i = d; i < 20 * 128; i += 128) sw[i] = Wsp[i];
    __syncthreads();
    const int n = blockIdx.x;
    if (n >= N) return;
    const int s = species[n];
    float acc = zrand[(size_t)s * 128 + d];
    const float* er = enc + (size_t)s * 20;
    #pragma unroll
    for (int j = 0; j < 20; ++j) acc += er[j] * sw[j * 128 + d];
    xi[(size_t)n * 128 + d] = acc;
}

// ---------------------------------------------------------------------------
// Per-edge radial + spherical basis (layer-invariant).
// rbY[e*32 + {0..15}] = rb,  rbY[e*32 + 16 + {0..8}] = Y.
// ---------------------------------------------------------------------------
__global__ __launch_bounds__(256) void basis_kernel(
    const float* __restrict__ vec, const float* __restrict__ dist,
    const float* __restrict__ sw, const float* __restrict__ gamma,
    float* __restrict__ rbY, int E)
{
    const int e = blockIdx.x * 256 + threadIdx.x;
    if (e >= E) return;
    const float BINOM[16] = {1.f, 15.f, 105.f, 455.f, 1365.f, 3003.f, 5005.f, 6435.f,
                             6435.f, 5005.f, 3003.f, 1365.f, 455.f, 105.f, 15.f, 1.f};
    const float S3 = 1.7320508075688772f;
    const float r  = dist[e];
    const float iv = __builtin_amdgcn_rcpf(r);
    const float x = vec[e * 3 + 0] * iv;
    const float y = vec[e * 3 + 1] * iv;
    const float z = vec[e * 3 + 2] * iv;
    float Y[9];
    Y[0] = 1.0f;  Y[1] = x;  Y[2] = y;  Y[3] = z;
    Y[4] = S3 * x * y;  Y[5] = S3 * y * z;  Y[6] = 0.5f * (3.0f * z * z - 1.0f);
    Y[7] = S3 * x * z;  Y[8] = 0.5f * S3 * (x * x - y * y);

    const float u = __expf(-gamma[0] * r);
    const float t = 1.0f - u;
    float up[16], tp[16];
    up[0] = 1.0f; tp[0] = 1.0f;
    #pragma unroll
    for (int k = 1; k < 16; ++k) { up[k] = up[k - 1] * u; tp[k] = tp[k - 1] * t; }
    const float s = sw[e];
    float* out = rbY + (size_t)e * 32;
    #pragma unroll
    for (int k = 0; k < 16; ++k) out[k] = BINOM[k] * up[k] * tp[15 - k] * s;
    #pragma unroll
    for (int j = 0; j < 9; ++j) out[16 + j] = Y[j];
}

// ---------------------------------------------------------------------------
// Edge messages + segment-sum (atomics). Per edge e, feature d:
//   A_l = sum_k rb[k] * W_l[k,d];  msg_{l,c} = feat_l[dst,d] * A_l * Y[c]
// ---------------------------------------------------------------------------
__global__ __launch_bounds__(256) void edge_msg_kernel(
    const int* __restrict__ esrc, const int* __restrict__ edst,
    const float* __restrict__ rbY,
    const float* __restrict__ sj, const float* __restrict__ pj,
    const float* __restrict__ dj,
    const float* __restrict__ Ws, const float* __restrict__ Wp,
    const float* __restrict__ Wd,
    float* __restrict__ si, float* __restrict__ pi, float* __restrict__ di,
    int E)
{
    __shared__ float sWs[NB * DIMF], sWp[NB * DIMF], sWd[NB * DIMF];
    __shared__ float sg[2][32];
    const int tid = threadIdx.x;
    for (int i = tid; i < NB * DIMF; i += 256) {
        sWs[i] = Ws[i]; sWp[i] = Wp[i]; sWd[i] = Wd[i];
    }
    const int sub = tid >> 7;
    const int d   = tid & 127;
    const long e  = (long)blockIdx.x * 2 + sub;
    const bool valid = (e < E);
    if (valid && d < 25) sg[sub][d] = rbY[e * 32 + d];
    __syncthreads();
    if (!valid) return;

    const int src = esrc[e];
    const int dst = edst[e];
    const float sjv = sj[(size_t)dst * 128 + d];
    const float pjv = pj[(size_t)dst * 128 + d];
    const float djv = dj[(size_t)dst * 128 + d];

    float As = 0.f, Ap = 0.f, Ad = 0.f;
    #pragma unroll
    for (int k = 0; k < NB; ++k) {
        const float rbk = sg[sub][k];
        As += rbk * sWs[k * DIMF + d];
        Ap += rbk * sWp[k * DIMF + d];
        Ad += rbk * sWd[k * DIMF + d];
    }
    const float* Y = &sg[sub][16];

    atomicAdd(&si[(size_t)src * 128 + d], sjv * As * Y[0]);
    #pragma unroll
    for (int c = 0; c < 3; ++c)
        atomicAdd(&pi[((size_t)src * 3 + c) * 128 + d], pjv * Ap * Y[1 + c]);
    #pragma unroll
    for (int c = 0; c < 5; ++c)
        atomicAdd(&di[((size_t)src * 5 + c) * 128 + d], djv * Ad * Y[4 + c]);
}

// ---------------------------------------------------------------------------
// agg[n,d] = c + si + sum_c P1*P2 + sum_c D1*D2
// ---------------------------------------------------------------------------
__global__ __launch_bounds__(256) void agg_kernel(
    const float* __restrict__ c, const float* __restrict__ si,
    const float* __restrict__ PP, const float* __restrict__ DD,
    float* __restrict__ agg, int total)
{
    const int i = blockIdx.x * 256 + threadIdx.x;
    if (i >= total) return;
    const int n = i >> 7;
    const int d = i & 127;
    float v = c[i] + si[i];
    #pragma unroll
    for (int cc = 0; cc < 3; ++cc) {
        const size_t b = ((size_t)n * 3 + cc) * 256;
        v += PP[b + d] * PP[b + 128 + d];
    }
    #pragma unroll
    for (int cc = 0; cc < 5; ++cc) {
        const size_t b = ((size_t)n * 5 + cc) * 256;
        v += DD[b + d] * DD[b + 128 + d];
    }
    agg[i] = v;
}

// ---------------------------------------------------------------------------
// Host orchestration
// ---------------------------------------------------------------------------
extern "C" void kernel_launch(void* const* d_in, const int* in_sizes, int n_in,
                              void* d_out, int out_size, void* d_ws, size_t ws_size,
                              hipStream_t stream)
{
    const int*   species   = (const int*)  d_in[0];
    const int*   edge_src  = (const int*)  d_in[1];
    const int*   edge_dst  = (const int*)  d_in[2];
    const float* vec       = (const float*)d_in[3];
    const float* distances = (const float*)d_in[4];
    const float* swtch     = (const float*)d_in[5];
    const float* enc_table = (const float*)d_in[6];
    const float* zrand     = (const float*)d_in[7];
    const float* W_species = (const float*)d_in[8];
    const float* gamma     = (const float*)d_in[9];
    const float* res_W1    = (const float*)d_in[10];
    const float* res_b1    = (const float*)d_in[11];
    const float* res_W2    = (const float*)d_in[12];
    const float* res_b2    = (const float*)d_in[13];
    const float* res_Wo    = (const float*)d_in[14];
    const float* res_bo    = (const float*)d_in[15];
    const float* Gs_W      = (const float*)d_in[16];
    const float* Gp_W      = (const float*)d_in[17];
    const float* Gd_W      = (const float*)d_in[18];
    const float* P12_W     = (const float*)d_in[19];
    const float* D12_W     = (const float*)d_in[20];

    const int N = in_sizes[0];
    const int E = in_sizes[1];

    // ---- workspace carve (fp32) ----
    float* w = (float*)d_ws;
    size_t off = 0;
    auto alloc = [&](size_t n) { float* p = w + off; off += n; return p; };
    float* rbY  = alloc((size_t)E * 32);
    float* xi   = alloc((size_t)N * 128);
    float* xt   = alloc((size_t)N * 128);
    float* cbuf = alloc((size_t)N * 128);
    float* sj   = alloc((size_t)N * 128);
    float* pj   = alloc((size_t)N * 128);
    float* dj   = alloc((size_t)N * 128);
    float* si   = alloc((size_t)N * 128);       // si, pi, di contiguous (one memset)
    float* pi   = alloc((size_t)N * 3 * 128);
    float* di   = alloc((size_t)N * 5 * 128);
    float* PP   = alloc((size_t)N * 3 * 256);
    float* DD   = alloc((size_t)N * 5 * 256);
    float* agg  = alloc((size_t)N * 128);
    float* lv   = alloc((size_t)N * 128);
    (void)ws_size;

    const size_t fusedShmem = (size_t)4 * 128 * 128 * 2 + 3 * 128 * 4; // 3 W + acts + biases
    const size_t gemmShmem  = (size_t)2 * 128 * 128 * 2;               // Nout=256 W

    hipMemsetAsync(d_out, 0, sizeof(float) * (size_t)out_size, stream);

    embed_kernel<<<N, 128, 0, stream>>>(species, enc_table, zrand, W_species, xi, N);
    basis_kernel<<<(E + 255) / 256, 256, 0, stream>>>(vec, distances, swtch, gamma, rbY, E);

    for (int l = 0; l < 3; ++l) {
        auto W1 = [&](int i) { return res_W1 + ((size_t)l * 8 + i) * 128 * 128; };
        auto b1 = [&](int i) { return res_b1 + ((size_t)l * 8 + i) * 128; };
        auto W2 = [&](int i) { return res_W2 + ((size_t)l * 8 + i) * 128 * 128; };
        auto b2 = [&](int i) { return res_b2 + ((size_t)l * 8 + i) * 128; };
        auto Wo = [&](int i) { return res_Wo + ((size_t)l * 8 + i) * 128 * 128; };
        auto bo = [&](int i) { return res_bo + ((size_t)l * 8 + i) * 128; };

        // fully fused resMLP: out (+)= silu(resblock(X+X2)) @ Wo + bo
        auto rm = [&](const float* X, const float* X2, int i, float* out, int accum) {
            resmlp_fused<<<N / 128, 256, fusedShmem, stream>>>(
                X, X2, W1(i), b1(i), W2(i), b2(i), Wo(i), bo(i),
                out, N, accum ? 2 : 0);
        };

        // xt = resblock(xi)   (fused kernel, resOnly mode)
        resmlp_fused<<<N / 128, 256, fusedShmem, stream>>>(
            xi, nullptr, W1(0), b1(0), W2(0), b2(0), Wo(0), bo(0),
            xt, N, 4);

        rm(xt, nullptr, 1, cbuf, 0);
        rm(xt, nullptr, 2, sj,   0);
        rm(xt, nullptr, 3, pj,   0);
        rm(xt, nullptr, 4, dj,   0);

        // message passing: si/pi/di = segment sums over edges
        hipMemsetAsync(si, 0, sizeof(float) * (size_t)N * 9 * 128, stream);
        edge_msg_kernel<<<(E + 1) / 2, 256, 0, stream>>>(
            edge_src, edge_dst, rbY, sj, pj, dj,
            Gs_W + (size_t)l * 16 * 128, Gp_W + (size_t)l * 16 * 128,
            Gd_W + (size_t)l * 16 * 128, si, pi, di, E);

        // PP = pi @ P12_W[l] (M = 3N, Nout = 256);  DD = di @ D12_W[l] (M = 5N)
        gemm128_wmma<<<(N * 3) / 64, 256, gemmShmem, stream>>>(
            pi, P12_W + (size_t)l * 128 * 256, PP, N * 3, 2);
        gemm128_wmma<<<(N * 5) / 64, 256, gemmShmem, stream>>>(
            di, D12_W + (size_t)l * 128 * 256, DD, N * 5, 2);

        agg_kernel<<<(N * 128 + 255) / 256, 256, 0, stream>>>(cbuf, si, PP, DD, agg,
                                                              N * 128);

        rm(agg, nullptr, 5, lv, 0);            // lv
        rm(xt,  lv,      6, xi, 0);            // xi = rm(xt + lv)
        rm(xi,  nullptr, 7, (float*)d_out, 1); // y += rm(xi)
    }
}
